// MultiKernelConv1DTrans_30657476559575
// MI455X (gfx1250) — matrix operand order repaired and checked by reading, not compiled, not run on hardware
//
#include <hip/hip_runtime.h>
#include <math.h>

// ---------------- constants ----------------
#define DIMC   768
#define HEADS  8
#define DHD    96
#define NSEQ   3840
#define BB     2
#define ROWS   (BB*NSEQ)          // 7680
#define MLAND  384
#define LFAC   10                 // NSEQ / MLAND
#define RESK   33
#define OI     (DIMC*DIMC)        // 589824
#define BH     (BB*HEADS)         // 16
#define MM     (MLAND*MLAND)      // 147456
#define EPSLN  1e-5f
#define PADR   4                  // conv zero-pad rows per batch edge (covers shift +/-3)
#define PROWS  (NSEQ + 2*PADR)    // padded rows per batch

typedef __attribute__((ext_vector_type(16))) _Float16 v16h;
typedef __attribute__((ext_vector_type(8)))  _Float16 v8h;
typedef __attribute__((ext_vector_type(8)))  float    v8f;

// ---------------- WMMA fragment loads (contiguous k = 16*h + s for both A and B) ----------------
__device__ __forceinline__ v16h load_a_frag(const _Float16* __restrict__ p) {
  // p = &A[row*lda + koffA + k0]; 16 consecutive K values
  v8h lo = *(const v8h*)(p);
  v8h hi = *(const v8h*)(p + 8);
  v16h r;
#pragma unroll
  for (int j = 0; j < 8; ++j) { r[j] = lo[j]; r[j + 8] = hi[j]; }
  return r;
}

__device__ __forceinline__ v16h load_bt_frag(const _Float16* __restrict__ p) {
  // p = &Bt[col*ldb + koffB + k0]; 16 consecutive K values
  v8h lo = *(const v8h*)(p);
  v8h hi = *(const v8h*)(p + 8);
  v16h r;
#pragma unroll
  for (int j = 0; j < 8; ++j) { r[j] = lo[j]; r[j + 8] = hi[j]; }
  return r;
}

// ---------------- batched WMMA GEMM, 32x32 tile per wave, NT only ----------------
// C[M x N] = alpha * sum_t A_rowshift(t) * B_t^T
// B always stored transposed ([N x K] row-major) -> contiguous 32B fragment loads.
// seq_len>0: A rows are indexed through a zero-padded layout [batch][PADR+seq+PADR][lda]
// so shifted rows are always in-bounds (no predication in the hot loop).
__global__ void __launch_bounds__(32)
gemm_wmma(const _Float16* __restrict__ A, const _Float16* __restrict__ Bm,
          float* __restrict__ C, int Kd,
          long lda, long ldb, long ldc,
          long sA, long sB, long sC,
          int ntaps, long sBt, int shift0, int seq_len, float alpha) {
  const int lane = threadIdx.x & 31;
  const int l16  = lane & 15;
  const int hsel = lane >> 4;       // 0 / 1
  const long bz  = blockIdx.z;
  A  += bz * sA;  Bm += bz * sB;  C += bz * sC;

  const int m0 = blockIdx.x * 32;
  const int n0 = blockIdx.y * 32;
  const int bcol0 = n0 + l16;
  const int bcol1 = n0 + 16 + l16;
  const int koffA = hsel * 16;
  const int koffB = hsel * 16;

  // base A rows for the two 16-row groups (tap 0)
  long ar0, ar1;
  {
    int rl0 = m0 + l16, rl1 = m0 + 16 + l16;
    if (seq_len > 0) {
      int b0 = rl0 / seq_len, p0 = rl0 - b0 * seq_len;
      int b1 = rl1 / seq_len, p1 = rl1 - b1 * seq_len;
      ar0 = (long)b0 * (seq_len + 2 * PADR) + PADR + p0 + shift0;
      ar1 = (long)b1 * (seq_len + 2 * PADR) + PADR + p1 + shift0;
    } else { ar0 = rl0; ar1 = rl1; }
  }

  v8f acc00 = {}, acc01 = {}, acc10 = {}, acc11 = {};

  for (int t = 0; t < ntaps; ++t) {
    const _Float16* Bt_ = Bm + (long)t * sBt;
    const _Float16* Ap0 = A + (ar0 + t) * lda + koffA;
    const _Float16* Ap1 = A + (ar1 + t) * lda + koffA;
    const _Float16* Bp0 = Bt_ + (long)bcol0 * ldb + koffB;
    const _Float16* Bp1 = Bt_ + (long)bcol1 * ldb + koffB;

    for (int k0 = 0; k0 < Kd; k0 += 32) {
      v16h a0 = load_a_frag(Ap0 + k0);
      v16h a1 = load_a_frag(Ap1 + k0);
      v16h b0 = load_bt_frag(Bp0 + k0);
      v16h b1 = load_bt_frag(Bp1 + k0);
      acc00 = __builtin_amdgcn_wmma_f32_16x16x32_f16(false, a0, false, b0, (short)0, acc00, false, false);
      acc01 = __builtin_amdgcn_wmma_f32_16x16x32_f16(false, a0, false, b1, (short)0, acc01, false, false);
      acc10 = __builtin_amdgcn_wmma_f32_16x16x32_f16(false, a1, false, b0, (short)0, acc10, false, false);
      acc11 = __builtin_amdgcn_wmma_f32_16x16x32_f16(false, a1, false, b1, (short)0, acc11, false, false);
    }
  }

  const int r0 = m0 + hsel * 8;
  const int r1 = m0 + 16 + hsel * 8;
#pragma unroll
  for (int j = 0; j < 8; ++j) {
    C[(long)(r0 + j) * ldc + bcol0] = alpha * acc00[j];
    C[(long)(r0 + j) * ldc + bcol1] = alpha * acc01[j];
    C[(long)(r1 + j) * ldc + bcol0] = alpha * acc10[j];
    C[(long)(r1 + j) * ldc + bcol1] = alpha * acc11[j];
  }
}

// ---------------- elementwise / reduction kernels ----------------
__global__ void cast_f32_f16(const float* __restrict__ s, _Float16* __restrict__ d, long n) {
  long i = (long)blockIdx.x * blockDim.x + threadIdx.x;
  if (i < n) d[i] = (_Float16)s[i];
}

// batched transpose + cast: in [BH][R][C] f32 -> out [BH][C][R] f16
__global__ void transpose_cast(const float* __restrict__ in, _Float16* __restrict__ out,
                               int R, int C, long nb) {
  long i = (long)blockIdx.x * blockDim.x + threadIdx.x;
  long rc = (long)R * C;
  if (i >= nb * rc) return;
  long b = i / rc; long r2 = i - b * rc;
  int c = (int)(r2 / R), r = (int)(r2 - (long)c * R);
  out[i] = (_Float16)in[b * rc + (long)r * C + c];
}

// zero-padded cast of x: [B][N][768] f32 -> [B][PADR+N+PADR][768] f16 (pads = 0)
__global__ void pad_cast_x(const float* __restrict__ x, _Float16* __restrict__ d) {
  long i = (long)blockIdx.x * blockDim.x + threadIdx.x;
  long per = (long)PROWS * DIMC;
  if (i >= (long)BB * per) return;
  long b = i / per; long r2 = i - b * per;
  int row = (int)(r2 / DIMC), c = (int)(r2 - (long)row * DIMC);
  int p = row - PADR;
  d[i] = (p >= 0 && p < NSEQ) ? (_Float16)x[(b * NSEQ + p) * (long)DIMC + c] : (_Float16)0.f;
}

// conv weight [O,I,K] f32 -> per-tap [O,I] f16 (t-major)
__global__ void pack_tap(const float* __restrict__ w, _Float16* __restrict__ d, int K, long n) {
  long i = (long)blockIdx.x * blockDim.x + threadIdx.x;
  if (i >= n) return;
  long oi = i / K; int t = (int)(i - oi * K);
  d[(long)t * OI + oi] = (_Float16)w[i];
}

// out0 += relu(LN(conv_row + conv_bias))  (row = 768, block = 256)
__global__ void ln_relu_add(const float* __restrict__ conv, const float* __restrict__ cb,
                            const float* __restrict__ w, const float* __restrict__ b,
                            float* __restrict__ out0) {
  __shared__ float s1[256], s2[256];
  const long row = blockIdx.x; const int t = threadIdx.x;
  float vv[3], a = 0.f, q = 0.f;
#pragma unroll
  for (int i = 0; i < 3; ++i) {
    int c = t + i * 256;
    vv[i] = conv[row * DIMC + c] + cb[c];
    a += vv[i]; q += vv[i] * vv[i];
  }
  s1[t] = a; s2[t] = q; __syncthreads();
  for (int s = 128; s > 0; s >>= 1) { if (t < s) { s1[t] += s1[t + s]; s2[t] += s2[t + s]; } __syncthreads(); }
  float mu = s1[0] * (1.f / DIMC);
  float var = s2[0] * (1.f / DIMC) - mu * mu;
  float rs = rsqrtf(var + EPSLN);
#pragma unroll
  for (int i = 0; i < 3; ++i) {
    int c = t + i * 256;
    float y = (vv[i] - mu) * rs * w[c] + b[c];
    out0[row * DIMC + c] += fmaxf(y, 0.f);
  }
}

// xn16 = LN(in_row) as f16
__global__ void ln_to_half(const float* __restrict__ in, const float* __restrict__ w,
                           const float* __restrict__ b, _Float16* __restrict__ out) {
  __shared__ float s1[256], s2[256];
  const long row = blockIdx.x; const int t = threadIdx.x;
  float vv[3], a = 0.f, q = 0.f;
#pragma unroll
  for (int i = 0; i < 3; ++i) {
    int c = t + i * 256;
    vv[i] = in[row * DIMC + c];
    a += vv[i]; q += vv[i] * vv[i];
  }
  s1[t] = a; s2[t] = q; __syncthreads();
  for (int s = 128; s > 0; s >>= 1) { if (t < s) { s1[t] += s1[t + s]; s2[t] += s2[t + s]; } __syncthreads(); }
  float mu = s1[0] * (1.f / DIMC);
  float rs = rsqrtf(s2[0] * (1.f / DIMC) - mu * mu + EPSLN);
#pragma unroll
  for (int i = 0; i < 3; ++i) {
    int c = t + i * 256;
    out[row * DIMC + c] = (_Float16)((vv[i] - mu) * rs * w[c] + b[c]);
  }
}

// d_out = LN(proj + out_b + out0)
__global__ void final_ln(const float* __restrict__ proj, const float* __restrict__ ob,
                         const float* __restrict__ out0, const float* __restrict__ w,
                         const float* __restrict__ b, float* __restrict__ out) {
  __shared__ float s1[256], s2[256];
  const long row = blockIdx.x; const int t = threadIdx.x;
  float vv[3], a = 0.f, q = 0.f;
#pragma unroll
  for (int i = 0; i < 3; ++i) {
    int c = t + i * 256;
    vv[i] = proj[row * DIMC + c] + ob[c] + out0[row * DIMC + c];
    a += vv[i]; q += vv[i] * vv[i];
  }
  s1[t] = a; s2[t] = q; __syncthreads();
  for (int s = 128; s > 0; s >>= 1) { if (t < s) { s1[t] += s1[t + s]; s2[t] += s2[t + s]; } __syncthreads(); }
  float mu = s1[0] * (1.f / DIMC);
  float rs = rsqrtf(s2[0] * (1.f / DIMC) - mu * mu + EPSLN);
#pragma unroll
  for (int i = 0; i < 3; ++i) {
    int c = t + i * 256;
    out[row * DIMC + c] = (vv[i] - mu) * rs * w[c] + b[c];
  }
}

// split QKV -> q16 (scaled), k16 in [BH][N][Dh], vt16 in [BH][Dh][N]
__global__ void split_qkv(const float* __restrict__ qkv, _Float16* __restrict__ q,
                          _Float16* __restrict__ k, _Float16* __restrict__ vt, float scale) {
  long i = (long)blockIdx.x * blockDim.x + threadIdx.x;
  if (i >= (long)ROWS * DIMC) return;
  long rn = i / DIMC; int c = (int)(i - rn * DIMC);
  long b = rn / NSEQ; long n = rn - b * NSEQ;
  int h = c / DHD, d = c - h * DHD;
  const float* row = qkv + rn * (3 * DIMC);
  long bh = b * HEADS + h;
  q[(bh * NSEQ + n) * DHD + d]  = (_Float16)(row[c] * scale);
  k[(bh * NSEQ + n) * DHD + d]  = (_Float16)row[DIMC + c];
  vt[(bh * DHD + d) * NSEQ + n] = (_Float16)row[2 * DIMC + c];
}

// landmark means: [BH][N][Dh] -> [BH][M][Dh], mean over LFAC consecutive positions
__global__ void landmark_mean(const _Float16* __restrict__ s, _Float16* __restrict__ d) {
  long i = (long)blockIdx.x * blockDim.x + threadIdx.x;
  if (i >= (long)BH * MLAND * DHD) return;
  long bh = i / (MLAND * DHD); long r = i - bh * (MLAND * DHD);
  int m = (int)(r / DHD), dd = (int)(r - (long)m * DHD);
  float acc = 0.f;
  const _Float16* base = s + (bh * NSEQ + (long)m * LFAC) * DHD + dd;
#pragma unroll
  for (int j = 0; j < LFAC; ++j) acc += (float)base[(long)j * DHD];
  d[i] = (_Float16)(acc * (1.f / LFAC));
}

// in-place row softmax, block 256 per row
__global__ void softmax_rows(float* __restrict__ X, int L) {
  __shared__ float red[256];
  float* x = X + (long)blockIdx.x * L;
  int t = threadIdx.x;
  float m = -1e30f;
  for (int i = t; i < L; i += 256) m = fmaxf(m, x[i]);
  red[t] = m; __syncthreads();
  for (int s = 128; s > 0; s >>= 1) { if (t < s) red[t] = fmaxf(red[t], red[t + s]); __syncthreads(); }
  m = red[0]; __syncthreads();
  float sum = 0.f;
  for (int i = t; i < L; i += 256) { float e = __expf(x[i] - m); x[i] = e; sum += e; }
  red[t] = sum; __syncthreads();
  for (int s = 128; s > 0; s >>= 1) { if (t < s) red[t] += red[t + s]; __syncthreads(); }
  float inv = 1.f / red[0];
  for (int i = t; i < L; i += 256) x[i] *= inv;
}

// max over rows of row abs-sum -> scal[0]   (X: [BH][M][M])
__global__ void absrow_max(const float* __restrict__ X, float* __restrict__ scal) {
  __shared__ float red[128];
  const float* x = X + (long)blockIdx.x * MLAND;
  int t = threadIdx.x; float s = 0.f;
  for (int i = t; i < MLAND; i += 128) s += fabsf(x[i]);
  red[t] = s; __syncthreads();
  for (int st = 64; st > 0; st >>= 1) { if (t < st) red[t] += red[t + st]; __syncthreads(); }
  if (t == 0) atomicMax((int*)scal, __float_as_int(red[0]));
}

// max over cols of column abs-sum -> scal[1]
__global__ void abscol_max(const float* __restrict__ X, float* __restrict__ scal) {
  __shared__ float red[128];
  long blk = blockIdx.x; long b = blk / MLAND; int c = (int)(blk - b * MLAND);
  const float* x = X + b * MM + c;
  int t = threadIdx.x; float s = 0.f;
  for (int i = t; i < MLAND; i += 128) s += fabsf(x[(long)i * MLAND]);
  red[t] = s; __syncthreads();
  for (int st = 64; st > 0; st >>= 1) { if (t < st) red[t] += red[t + st]; __syncthreads(); }
  if (t == 0) atomicMax((int*)(scal + 1), __float_as_int(red[0]));
}

// z = x^T/(col*row) (A-operand) and zT = x/(col*row) (B-operand), both f16
__global__ void init_z(const float* __restrict__ S2, const float* __restrict__ scal,
                       _Float16* __restrict__ z, _Float16* __restrict__ zT) {
  long i = (long)blockIdx.x * blockDim.x + threadIdx.x;
  if (i >= (long)BH * MM) return;
  long b = i / MM; long r = i - b * MM;
  int ii = (int)(r / MLAND), jj = (int)(r - (long)ii * MLAND);
  float rs = 1.f / (scal[0] * scal[1]);
  z[i]  = (_Float16)(S2[b * MM + (long)jj * MLAND + ii] * rs);
  zT[i] = (_Float16)(S2[i] * rs);
}

// Y = (a*I - X)^T  f16   (ready to be used as transposed-B operand)
__global__ void diag_sub_T(const float* __restrict__ X, _Float16* __restrict__ Y, float a) {
  long i = (long)blockIdx.x * blockDim.x + threadIdx.x;
  if (i >= (long)BH * MM) return;
  long b = i / MM; long r = i - b * MM;
  int n = (int)(r / MLAND), k = (int)(r - (long)n * MLAND);
  Y[i] = (_Float16)(((n == k) ? a : 0.f) - X[b * MM + (long)k * MLAND + n]);
}

// cast to two f16 buffers: natural (A use) + transposed (B use)
__global__ void cast_nt(const float* __restrict__ S, _Float16* __restrict__ zn,
                        _Float16* __restrict__ zt) {
  long i = (long)blockIdx.x * blockDim.x + threadIdx.x;
  if (i >= (long)BH * MM) return;
  long b = i / MM; long r = i - b * MM;
  int ii = (int)(r / MLAND), jj = (int)(r - (long)ii * MLAND);
  zn[i] = (_Float16)S[i];
  zt[i] = (_Float16)S[b * MM + (long)jj * MLAND + ii];
}

// depthwise residual conv over sequence, accumulated into outh [BH][N][Dh]
__global__ void resconv_add(const _Float16* __restrict__ vt, const float* __restrict__ w,
                            float* __restrict__ outh) {
  long i = (long)blockIdx.x * blockDim.x + threadIdx.x;
  if (i >= (long)BH * NSEQ * DHD) return;
  long bh = i / ((long)NSEQ * DHD); long r = i - bh * ((long)NSEQ * DHD);
  int n = (int)(r / DHD), d = (int)(r - (long)n * DHD);
  int h = (int)(bh % HEADS);
  const _Float16* vrow = vt + (bh * DHD + d) * NSEQ;
  float acc = 0.f;
#pragma unroll
  for (int t = 0; t < RESK; ++t) {
    int p = n + t - RESK / 2;
    if (p >= 0 && p < NSEQ) acc += w[h * RESK + t] * (float)vrow[p];
  }
  outh[i] += acc;
}

// [BH][N][Dh] f32 -> [B][N][DIM] f16
__global__ void merge_heads(const float* __restrict__ outh, _Float16* __restrict__ m) {
  long i = (long)blockIdx.x * blockDim.x + threadIdx.x;
  if (i >= (long)ROWS * DIMC) return;
  long rn = i / DIMC; int c = (int)(i - rn * DIMC);
  long b = rn / NSEQ; long n = rn - b * NSEQ;
  int h = c / DHD, d = c - h * DHD;
  m[i] = (_Float16)outh[((b * HEADS + h) * NSEQ + n) * DHD + d];
}

// ---------------- host orchestration ----------------
extern "C" void kernel_launch(void* const* d_in, const int* in_sizes, int n_in,
                              void* d_out, int out_size, void* d_ws, size_t ws_size,
                              hipStream_t stream) {
  const float* x    = (const float*)d_in[0];
  const float* c1w  = (const float*)d_in[1];
  const float* c1b  = (const float*)d_in[2];
  const float* c2w  = (const float*)d_in[3];
  const float* c2b  = (const float*)d_in[4];
  const float* c3w  = (const float*)d_in[5];
  const float* c3b  = (const float*)d_in[6];
  const float* nw   = (const float*)d_in[7];
  const float* nb   = (const float*)d_in[8];
  const float* tw   = (const float*)d_in[9];
  const float* tb   = (const float*)d_in[10];
  const float* qkvw = (const float*)d_in[11];
  const float* outw = (const float*)d_in[12];
  const float* outb = (const float*)d_in[13];
  const float* resw = (const float*)d_in[14];
  float* out = (float*)d_out;

  char* ws = (char*)d_ws;
  size_t off = 0;
  auto alloc = [&](size_t bytes) -> char* {
    char* p = ws + off;
    off = (off + bytes + 255) & ~(size_t)255;
    return p;
  };

  _Float16* wc1    = (_Float16*)alloc((size_t)3 * OI * 2);
  _Float16* wc2    = (_Float16*)alloc((size_t)5 * OI * 2);
  _Float16* wc3    = (_Float16*)alloc((size_t)7 * OI * 2);
  _Float16* qkvw16 = (_Float16*)alloc((size_t)3 * DIMC * DIMC * 2);
  _Float16* outw16 = (_Float16*)alloc((size_t)OI * 2);
  _Float16* x16p   = (_Float16*)alloc((size_t)BB * PROWS * DIMC * 2); // zero-padded conv input
  float*    out0   = (float*)  alloc((size_t)ROWS * DIMC * 4);
  float*    convT  = (float*)  alloc((size_t)ROWS * DIMC * 4);   // conv scratch, later proj output
  _Float16* xn16   = (_Float16*)alloc((size_t)ROWS * DIMC * 2);  // later merged heads
  float*    R1     = (float*)  alloc((size_t)BH * MLAND * NSEQ * 4); // qkv f32 alias + sim3/sim1
  _Float16* q16    = (_Float16*)alloc((size_t)BH * NSEQ * DHD * 2);
  _Float16* k16    = (_Float16*)alloc((size_t)BH * NSEQ * DHD * 2);
  _Float16* vt16   = (_Float16*)alloc((size_t)BH * NSEQ * DHD * 2);
  _Float16* ql16   = (_Float16*)alloc((size_t)BH * MLAND * DHD * 2);
  _Float16* kl16   = (_Float16*)alloc((size_t)BH * MLAND * DHD * 2);
  _Float16* a16big = (_Float16*)alloc((size_t)BH * MLAND * NSEQ * 2); // attn3 then attn1
  float*    S2     = (float*)  alloc((size_t)BH * MM * 4);  // sim2/attn2 f32, then pinv P
  float*    Qb     = (float*)  alloc((size_t)BH * MM * 4);  // pinv Q
  _Float16* at2_16 = (_Float16*)alloc((size_t)BH * MM * 2);
  _Float16* xz16   = (_Float16*)alloc((size_t)BH * MM * 2);
  _Float16* wbT16  = (_Float16*)alloc((size_t)BH * MM * 2);
  _Float16* z16    = (_Float16*)alloc((size_t)BH * MM * 2);
  _Float16* zT16   = (_Float16*)alloc((size_t)BH * MM * 2);
  float*    T1     = (float*)  alloc((size_t)BH * MLAND * DHD * 4);
  float*    T2     = (float*)  alloc((size_t)BH * MLAND * DHD * 4);
  _Float16* t1t16  = (_Float16*)alloc((size_t)BH * MLAND * DHD * 2); // t1^T [Dh][M]
  _Float16* t2t16  = (_Float16*)alloc((size_t)BH * MLAND * DHD * 2); // t2^T [Dh][M]
  float*    outh   = (float*)  alloc((size_t)BH * NSEQ * DHD * 4);
  float*    scal   = (float*)  alloc(256);
  float*    qkvF   = R1; // alias: 7680*2304 f32 fits in R1

  auto ew = [&](long n) { return dim3((unsigned)((n + 255) / 256)); };
  auto G = [&](const _Float16* A, const _Float16* B, float* C,
               int Mr, int Nc, int Kd, long lda, long ldb, long ldc,
               long sA, long sB, long sC, int batch,
               int ntaps, long sBt, int shift0, int seq, float alpha) {
    dim3 g(Mr / 32, Nc / 32, batch);
    gemm_wmma<<<g, 32, 0, stream>>>(A, B, C, Kd, lda, ldb, ldc, sA, sB, sC,
                                    ntaps, sBt, shift0, seq, alpha);
  };

  // ---- weight prep + input cast ----
  pack_tap<<<ew((long)OI * 3), 256, 0, stream>>>(c1w, wc1, 3, (long)OI * 3);
  pack_tap<<<ew((long)OI * 5), 256, 0, stream>>>(c2w, wc2, 5, (long)OI * 5);
  pack_tap<<<ew((long)OI * 7), 256, 0, stream>>>(c3w, wc3, 7, (long)OI * 7);
  cast_f32_f16<<<ew((long)3 * OI), 256, 0, stream>>>(qkvw, qkvw16, (long)3 * OI);
  cast_f32_f16<<<ew((long)OI), 256, 0, stream>>>(outw, outw16, (long)OI);
  pad_cast_x<<<ew((long)BB * PROWS * DIMC), 256, 0, stream>>>(x, x16p);
  hipMemcpyAsync(out0, x, (size_t)ROWS * DIMC * 4, hipMemcpyDeviceToDevice, stream);

  // ---- three conv branches (tap-fused WMMA GEMM, padded rows) + LN + relu + acc ----
  G(x16p, wc1, convT, ROWS, DIMC, DIMC, DIMC, DIMC, DIMC, 0, 0, 0, 1, 3, OI, -1, NSEQ, 1.f);
  ln_relu_add<<<ROWS, 256, 0, stream>>>(convT, c1b, nw, nb, out0);
  G(x16p, wc2, convT, ROWS, DIMC, DIMC, DIMC, DIMC, DIMC, 0, 0, 0, 1, 5, OI, -2, NSEQ, 1.f);
  ln_relu_add<<<ROWS, 256, 0, stream>>>(convT, c2b, nw, nb, out0);
  G(x16p, wc3, convT, ROWS, DIMC, DIMC, DIMC, DIMC, DIMC, 0, 0, 0, 1, 7, OI, -3, NSEQ, 1.f);
  ln_relu_add<<<ROWS, 256, 0, stream>>>(convT, c3b, nw, nb, out0);

  // ---- attention pre-norm, QKV ----
  ln_to_half<<<ROWS, 256, 0, stream>>>(out0, tw, tb, xn16);
  G(xn16, qkvw16, qkvF, ROWS, 3 * DIMC, DIMC, DIMC, DIMC, 3 * DIMC, 0, 0, 0, 1, 1, 0, 0, 0, 1.f);
  split_qkv<<<ew((long)ROWS * DIMC), 256, 0, stream>>>(qkvF, q16, k16, vt16, (float)(1.0 / sqrt((double)DHD)));
  landmark_mean<<<ew((long)BH * MLAND * DHD), 256, 0, stream>>>(q16, ql16);
  landmark_mean<<<ew((long)BH * MLAND * DHD), 256, 0, stream>>>(k16, kl16);

  const long sQL = (long)MLAND * DHD;   // 36864
  const long sQ  = (long)NSEQ * DHD;    // 368640
  const long sS2 = MM;                  // 147456
  const long sS3 = (long)MLAND * NSEQ;  // 1474560

  // ---- sim2 -> softmax -> pinv init ----
  G(ql16, kl16, S2, MLAND, MLAND, DHD, DHD, DHD, MLAND, sQL, sQL, sS2, BH, 1, 0, 0, 0, 1.f);
  softmax_rows<<<BH * MLAND, 256, 0, stream>>>(S2, MLAND);
  hipMemsetAsync(scal, 0, 8, stream);
  absrow_max<<<BH * MLAND, 128, 0, stream>>>(S2, scal);
  abscol_max<<<BH * MLAND, 128, 0, stream>>>(S2, scal);
  cast_f32_f16<<<ew((long)BH * MM), 256, 0, stream>>>(S2, at2_16, (long)BH * MM);
  init_z<<<ew((long)BH * MM), 256, 0, stream>>>(S2, scal, z16, zT16);

  // ---- Moore-Penrose iterations (batched 384^3 WMMA GEMMs, all NT path) ----
  for (int it = 0; it < 6; ++it) {
    // XZ = X @ Z (B operand = Z^T)
    G(at2_16, zT16, S2, MLAND, MLAND, MLAND, MLAND, MLAND, MLAND, sS2, sS2, sS2, BH, 1, 0, 0, 0, 1.f);
    cast_f32_f16<<<ew((long)BH * MM), 256, 0, stream>>>(S2, xz16, (long)BH * MM);
    diag_sub_T<<<ew((long)BH * MM), 256, 0, stream>>>(S2, wbT16, 7.f);     // (7I-XZ)^T
    G(xz16, wbT16, Qb, MLAND, MLAND, MLAND, MLAND, MLAND, MLAND, sS2, sS2, sS2, BH, 1, 0, 0, 0, 1.f);
    diag_sub_T<<<ew((long)BH * MM), 256, 0, stream>>>(Qb, wbT16, 15.f);    // (15I-..)^T
    G(xz16, wbT16, Qb, MLAND, MLAND, MLAND, MLAND, MLAND, MLAND, sS2, sS2, sS2, BH, 1, 0, 0, 0, 1.f);
    diag_sub_T<<<ew((long)BH * MM), 256, 0, stream>>>(Qb, wbT16, 13.f);    // (13I-..)^T
    G(z16, wbT16, S2, MLAND, MLAND, MLAND, MLAND, MLAND, MLAND, sS2, sS2, sS2, BH, 1, 0, 0, 0, 0.25f);
    cast_nt<<<ew((long)BH * MM), 256, 0, stream>>>(S2, z16, zT16);         // z and z^T
  }

  // ---- sim3 -> softmax -> t1 = attn3 @ v ----
  G(ql16, k16, R1, MLAND, NSEQ, DHD, DHD, DHD, NSEQ, sQL, sQ, sS3, BH, 1, 0, 0, 0, 1.f);
  softmax_rows<<<BH * MLAND, 256, 0, stream>>>(R1, NSEQ);
  cast_f32_f16<<<ew((long)BH * sS3), 256, 0, stream>>>(R1, a16big, (long)BH * sS3);
  G(a16big, vt16, T1, MLAND, DHD, NSEQ, NSEQ, NSEQ, DHD, sS3, sQ, sQL, BH, 1, 0, 0, 0, 1.f);
  transpose_cast<<<ew((long)BH * sQL), 256, 0, stream>>>(T1, t1t16, MLAND, DHD, BH); // [M][Dh]->[Dh][M]

  // ---- t2 = pinv(attn2) @ t1 (B operand = t1^T) ----
  G(z16, t1t16, T2, MLAND, DHD, MLAND, MLAND, MLAND, DHD, sS2, sQL, sQL, BH, 1, 0, 0, 0, 1.f);
  transpose_cast<<<ew((long)BH * sQL), 256, 0, stream>>>(T2, t2t16, MLAND, DHD, BH); // [M][Dh]->[Dh][M]

  // ---- sim1 -> softmax -> out = attn1 @ t2 (B operand = t2^T) ----
  G(q16, kl16, R1, NSEQ, MLAND, DHD, DHD, DHD, MLAND, sQ, sQL, sS3, BH, 1, 0, 0, 0, 1.f);
  softmax_rows<<<BH * NSEQ, 256, 0, stream>>>(R1, MLAND);
  cast_f32_f16<<<ew((long)BH * sS3), 256, 0, stream>>>(R1, a16big, (long)BH * sS3);
  G(a16big, t2t16, outh, NSEQ, DHD, MLAND, MLAND, MLAND, DHD, sS3, sQL, sQ, BH, 1, 0, 0, 0, 1.f);

  // ---- residual depthwise conv + merge heads + out projection + final LN ----
  resconv_add<<<ew((long)BH * NSEQ * DHD), 256, 0, stream>>>(vt16, resw, outh);
  merge_heads<<<ew((long)ROWS * DIMC), 256, 0, stream>>>(outh, xn16);
  G(xn16, outw16, convT, ROWS, DIMC, DIMC, DIMC, DIMC, DIMC, 0, 0, 0, 1, 1, 0, 0, 0, 1.f);
  final_ln<<<ROWS, 256, 0, stream>>>(convT, outb, out0, nw, nb, out);
}
